// DGPLoss_44263932952734
// MI455X (gfx1250) — matrix operand order repaired and checked by reading, not compiled, not run on hardware
//
#include <hip/hip_runtime.h>
#include <hip/hip_bf16.h>

// DGP loss for MI455X (gfx1250, wave32).
// sd2 = ||a||^2 + ||b||^2 - 2 a.b ; cross-dots via v_wmma_f32_16x16x32_bf16
// banded local-correlation GEMM staged through LDS (bf16, padded stride).
// Round 3: fully branchless epilogue (select on exp *input*, exp(-big) == +0).

typedef __attribute__((ext_vector_type(16))) __bf16 v16bf;
typedef __attribute__((ext_vector_type(8)))  float  v8f;

#define IMG_W   384
#define IMG_C   128
#define OUT_W   380
#define TILE_X  12      // valid centers per x-tile (A rows 0..11 used of 16)
#define TILE_Y  8       // output rows per block
#define BLK_PIX_ROWS 12 // TILE_Y + 4 halo rows
#define SEG_STRIDE 136  // 128 ch + 8 pad (bf16 units) -> 272 B, 16B aligned

union Frag {
    v16bf bf;
    uint4 q[2];
};

static __device__ __forceinline__ unsigned short f2bf(float f) {
    unsigned u = __float_as_uint(f);
    u += 0x7fffu + ((u >> 16) & 1u);      // round-to-nearest-even
    return (unsigned short)(u >> 16);
}

__global__ __launch_bounds__(128)
void dgp_loss_kernel(const float* __restrict__ seg,
                     const float* __restrict__ dep,
                     float* __restrict__ ws) {
    __shared__ unsigned short s_seg[BLK_PIX_ROWS * 16 * SEG_STRIDE]; // 52,224 B
    __shared__ float s_n[BLK_PIX_ROWS * 16];
    __shared__ float s_dep[BLK_PIX_ROWS * 16];

    const int t      = threadIdx.x;
    const int b      = blockIdx.z;
    const int x_base = blockIdx.x * TILE_X;     // output-x of center p=0
    const int y0     = blockIdx.y * TILE_Y;     // output-y of row oy=0
    const int valid_p = min(TILE_X, OUT_W - x_base);
    const int valid_y = min(TILE_Y, OUT_W - y0);

    // ---- Phase 1: global f32 -> LDS bf16 (pixel-major [pix][ch]), zero-fill OOB
    // work: 64 channel-pairs x 12 rows x 4 float4-cols = 3072 items, 24/thread
    for (int it = t; it < 64 * BLK_PIX_ROWS * 4; it += 128) {
        const int c4    = it & 3;
        const int row   = (it >> 2) % BLK_PIX_ROWS;
        const int ch    = (it / (4 * BLK_PIX_ROWS)) * 2;
        const int row_g = y0 + row;
        const int col0  = x_base + c4 * 4;
        float a0[4], a1[4];
        if (row_g < IMG_W) {
            const float* base0 = seg + (((size_t)(b * IMG_C + ch)) * IMG_W + row_g) * IMG_W;
            const float* base1 = base0 + (size_t)IMG_W * IMG_W;
            if (col0 + 3 < IMG_W) {
                float4 va = *(const float4*)(base0 + col0);
                float4 vb = *(const float4*)(base1 + col0);
                a0[0] = va.x; a0[1] = va.y; a0[2] = va.z; a0[3] = va.w;
                a1[0] = vb.x; a1[1] = vb.y; a1[2] = vb.z; a1[3] = vb.w;
            } else {
                #pragma unroll
                for (int e = 0; e < 4; ++e) {
                    const int cg = col0 + e;
                    a0[e] = (cg < IMG_W) ? base0[cg] : 0.0f;
                    a1[e] = (cg < IMG_W) ? base1[cg] : 0.0f;
                }
            }
        } else {
            #pragma unroll
            for (int e = 0; e < 4; ++e) { a0[e] = 0.0f; a1[e] = 0.0f; }
        }
        #pragma unroll
        for (int e = 0; e < 4; ++e) {
            const int pix = row * 16 + c4 * 4 + e;
            const unsigned packed = (unsigned)f2bf(a0[e]) | ((unsigned)f2bf(a1[e]) << 16);
            *(unsigned*)(s_seg + pix * SEG_STRIDE + ch) = packed; // 4B aligned
        }
    }
    __syncthreads();

    // ---- Phase 2: per-pixel squared norms (from the bf16-rounded values) + depth tile
    for (int p = t; p < BLK_PIX_ROWS * 16; p += 128) {
        const unsigned short* rowp = s_seg + p * SEG_STRIDE;
        float nacc = 0.0f;
        #pragma unroll
        for (int kk = 0; kk < 16; ++kk) {
            uint4 v = *(const uint4*)(rowp + kk * 8);
            unsigned u[4] = {v.x, v.y, v.z, v.w};
            #pragma unroll
            for (int e = 0; e < 4; ++e) {
                float f0 = __uint_as_float(u[e] << 16);
                float f1 = __uint_as_float(u[e] & 0xffff0000u);
                nacc += f0 * f0 + f1 * f1;
            }
        }
        s_n[p] = nacc;
    }
    for (int i = t; i < BLK_PIX_ROWS * 16; i += 128) {
        const int row = i >> 4, col = i & 15;
        const int rg = y0 + row, cg = x_base + col;
        s_dep[i] = (rg < IMG_W && cg < IMG_W)
                 ? dep[((size_t)b * IMG_W + rg) * IMG_W + cg] : 0.0f;
    }
    __syncthreads();

    // ---- Phase 3: banded correlation GEMM per wave (full EXEC at every WMMA)
    const int wave = t >> 5, lane = t & 31;
    const int half = lane >> 4, l15 = lane & 15;
    float lsum = 0.0f, lcnt = 0.0f;

    for (int oy = wave; oy < TILE_Y; oy += 4) {        // uniform per wave
        if (oy >= valid_y) continue;                   // uniform: EXEC stays full

        // A: 16 rows = centers p=0..11 (rows 12..15 dummy, never read from D)
        // 16-bit A 16x32 layout: lane<16 holds K {0..7,16..23}, lane>=16 {8..15,24..31}
        Frag A[4];
        const int pixA = (oy + 2) * 16 + min(l15 + 2, 15);
        #pragma unroll
        for (int kc = 0; kc < 4; ++kc) {
            const unsigned short* pa = s_seg + pixA * SEG_STRIDE + kc * 32 + half * 8;
            A[kc].q[0] = *(const uint4*)pa;        // K k0 .. k0+7  (+halfsel)
            A[kc].q[1] = *(const uint4*)(pa + 16); // K k0+16..k0+23(+halfsel)
        }

        // Hoisted per-lane broadcasts: center terms depend only on r (p),
        // neighbor terms depend only on di (q = l15 fixed per lane).
        const int rowC = (oy + 2) * 16;
        float n_c[8], d_c[8];
        bool  pv[8], pv2[8];
        #pragma unroll
        for (int r = 0; r < 8; ++r) {
            const int p = half ? (r + 8) : r;
            const int pc = min(p + 2, 15);         // clamp (entries masked anyway)
            n_c[r] = s_n[rowC + pc];
            d_c[r] = s_dep[rowC + pc];
            const int j = l15 - p;                 // neighbor j-offset
            pv[r]  = (p < valid_p) && (j >= 0) && (j <= 4);
            pv2[r] = pv[r] && (j != 2);            // mask used when di == 2 (center)
        }

        #pragma unroll
        for (int di = 0; di < 5; ++di) {
            v8f acc = {0.f, 0.f, 0.f, 0.f, 0.f, 0.f, 0.f, 0.f};
            const int pixB = (oy + di) * 16 + l15;
            // B 32x16 layout: lanes<16 hold K {0..7,16..23}, lanes>=16 {8..15,24..31}
            const unsigned short* pb0 = s_seg + pixB * SEG_STRIDE + half * 8;
            #pragma unroll
            for (int kc = 0; kc < 4; ++kc) {
                Frag Bf;
                const unsigned short* pb = pb0 + kc * 32;
                Bf.q[0] = *(const uint4*)pb;
                Bf.q[1] = *(const uint4*)(pb + 16);
                acc = __builtin_amdgcn_wmma_f32_16x16x32_bf16(
                          false, A[kc].bf, false, Bf.bf,
                          (short)0, acc, false, false);
            }

            // Fully branchless epilogue on owned D entries:
            // VGPR r -> M (= center p), lane -> N (= q = l15).
            // Select on the exp *input*: exp(-3e38) underflows to exactly +0,
            // so masked entries add nothing and no control flow is generated.
            const float n_o   = s_n[(oy + di) * 16 + l15];
            const float dep_o = s_dep[(oy + di) * 16 + l15];
            const bool  dep_ok = dep_o > 1e-8f;
            #pragma unroll
            for (int r = 0; r < 8; ++r) {
                const float dot = acc[r];
                const float sd2 = fmaxf(n_c[r] + n_o - 2.0f * dot, 0.0f);
                const float ddv = fabsf(d_c[r] - dep_o);
                const bool  m = (di == 2 ? pv2[r] : pv[r]) &&
                                dep_ok && (ddv > 1e-8f) && (sd2 > 1e-16f);
                // exp(-dd/TAU)*exp(-sd2) = exp(-(dd*0.1 + sd2))
                const float arg = m ? (-(ddv * 0.1f) - sd2) : -3.0e38f;
                lsum += __expf(arg);           // +0.0 when masked
                lcnt += m ? 1.0f : 0.0f;
            }
        }
    }

    // ---- wave32 reduction, then 2 global float atomics
    #pragma unroll
    for (int off = 16; off > 0; off >>= 1) {
        lsum += __shfl_xor(lsum, off, 32);
        lcnt += __shfl_xor(lcnt, off, 32);
    }
    if (lane == 0) {
        atomicAdd(&ws[0], lsum);
        atomicAdd(&ws[1], lcnt);
    }
}

__global__ void dgp_finalize_kernel(const float* __restrict__ ws,
                                    float* __restrict__ out) {
    out[0] = ws[0] / ws[1] * 1.0f;  // SCALE = 1.0
}

extern "C" void kernel_launch(void* const* d_in, const int* in_sizes, int n_in,
                              void* d_out, int out_size, void* d_ws, size_t ws_size,
                              hipStream_t stream) {
    const float* seg = (const float*)d_in[0];   // (4,128,384,384) f32
    const float* dep = (const float*)d_in[1];   // (4,1,384,384)   f32
    float* ws  = (float*)d_ws;
    float* out = (float*)d_out;

    hipMemsetAsync(d_ws, 0, 2 * sizeof(float), stream);

    dim3 grid((OUT_W + TILE_X - 1) / TILE_X,    // 32
              (OUT_W + TILE_Y - 1) / TILE_Y,    // 48
              4);                               // batch
    dgp_loss_kernel<<<grid, 128, 0, stream>>>(seg, dep, ws);
    dgp_finalize_kernel<<<1, 1, 0, stream>>>(ws, out);
}
